// AdaptiveGeometryModule_39797166965045
// MI455X (gfx1250) — matrix operand order repaired and checked by reading, not compiled, not run on hardware
//
#include <hip/hip_runtime.h>
#include <hip/hip_bf16.h>
#include <math.h>
#include <stdint.h>

// ---------------------------------------------------------------------------
// AdaptiveGeometryModule — CDNA5 (gfx1250) implementation.
//   Seq-len-1 attention is the identity => refined = balanced @ W_eff + b_eff
//   with W_eff = Wv @ proj_w precomputed. GEMMs run on v_wmma_f32_16x16x32_bf16;
//   the dominant GEMM streams W_eff^T K-panels via the Tensor Data Mover into
//   a double-buffered LDS panel (TENSORcnt-synchronized), with TDM LDS padding
//   chosen so b128 fragment reads are 16B-aligned and bank-conflict-free.
// ---------------------------------------------------------------------------

typedef __bf16 bf16;
typedef __attribute__((ext_vector_type(16))) __bf16 v16bf;
typedef __attribute__((ext_vector_type(8)))  __bf16 v8bf;
typedef __attribute__((ext_vector_type(8)))  float  v8f;
typedef __attribute__((ext_vector_type(4)))  unsigned int u32x4;
typedef __attribute__((ext_vector_type(8)))  int i32x8;
typedef __attribute__((ext_vector_type(4)))  int i32x4;

union BF16x16 { v16bf v; v8bf h[2]; };

#define Bn 16384
#define Dn 512
#define An 10
#define HALF_D 256

#define ROWS 32        // rows per workgroup tile
#define LDK  520       // padded bf16 row stride (512 + 8)
#define LDH  264       // padded f32 row stride for H1 (256 + 8)
#define PANROW 80      // TDM panel row stride in bytes: 64B data + 16B pad

// Workspace layout (in d_ws)
struct WS {
    float center_sum;
    float div_sum;
    float pad[14];
    float anch[An * Dn];           // anchors + pos_embedding (fp32)
    float anchnorm2[16];
    float b_eff[Dn];               // bv @ proj_w + proj_b
    bf16  Wt[Dn * Dn];             // W_eff^T  (N-major: Wt[n*512 + k]), bf16
    bf16  w1T[HALF_D * Dn];        // w1^T     (N-major: w1T[n*512 + k]), bf16
};

// ---------------------------------------------------------------------------
// TDM: load one 512(N) x 32(K) bf16 panel of Wt (N-major, dim0-stride 512)
// into LDS at lds_off, padding +4 DWORDs after every 16 DWORDs stored
// (=> one 64B row + 16B pad = 80B row stride in LDS).
// D# packing per CDNA5 ISA.
// This toolchain carries the 6-arg builtin (extra i32x8, zero-filled).
// ---------------------------------------------------------------------------
__device__ __forceinline__ void tdm_load_panel(const bf16* gbase, unsigned lds_off) {
    const unsigned long long ga = (unsigned long long)(uintptr_t)gbase;
    u32x4 g0;
    g0[0] = 1u;                                             // count=1, user desc
    g0[1] = lds_off;                                        // lds_addr (bytes)
    g0[2] = (unsigned)(ga & 0xffffffffu);                   // global_addr[31:0]
    g0[3] = (unsigned)((ga >> 32) & 0x01ffffffu)            // global_addr[56:32]
          | 0x80000000u;                                    // type=2 ("image")
    i32x8 g1;
    g1[0] = (1 << 16)        // data_size = 2 bytes
          | (1 << 20)        // pad_enable
          | (3 << 22)        // pad_interval: 16 DWORDs
          | (3 << 25);       // pad_amount:   4 DWORDs
    g1[1] = (int)(512u << 16);   // tensor_dim0[15:0]=512 @ bits[63:48]
    g1[2] = (int)(512u << 16);   // tensor_dim0 hi=0; tensor_dim1[15:0]=512
    g1[3] = (int)(32u  << 16);   // tensor_dim1 hi=0; tile_dim0=32
    g1[4] = 512;                 // tile_dim1=512; tile_dim2=0
    g1[5] = 512;                 // tensor_dim0_stride[31:0]=512
    g1[6] = (int)(512u << 16);   // dim0_stride hi=0; tensor_dim1_stride[15:0]=512
    g1[7] = 0;                   // tensor_dim1_stride hi=0
    const i32x4 z4 = {0, 0, 0, 0};
    const i32x8 z8 = {0, 0, 0, 0, 0, 0, 0, 0};
    __builtin_amdgcn_tensor_load_to_lds(g0, g1, z4, z4, z8, 0);
}

// ---------------------------------------------------------------------------
// Kernel 1: tiny prep (anch, anchor norms, b_eff, diversity loss, zero accum)
// ---------------------------------------------------------------------------
__global__ void k_prep(const float* __restrict__ anchors,
                       const float* __restrict__ pos,
                       const float* __restrict__ qkv_b,
                       const float* __restrict__ proj_w,
                       const float* __restrict__ proj_b,
                       WS* __restrict__ ws) {
    const int t = threadIdx.x;
    if (t == 0) ws->center_sum = 0.0f;

    for (int e = t; e < An * Dn; e += 256)
        ws->anch[e] = anchors[e] + pos[e];
    __syncthreads();

    if (t < An) {
        float s = 0.0f;
        for (int k = 0; k < Dn; ++k) { float v = ws->anch[t * Dn + k]; s += v * v; }
        ws->anchnorm2[t] = s;
    }

    for (int n = t; n < Dn; n += 256) {
        float s = proj_b[n];
        for (int k = 0; k < Dn; ++k)
            s += qkv_b[2 * Dn + k] * proj_w[(size_t)k * Dn + n];
        ws->b_eff[n] = s;
    }

    __shared__ float divacc;
    if (t == 0) divacc = 0.0f;
    __syncthreads();
    if (t < (An * (An - 1)) / 2) {
        int c = t, i = 0, j = 1;
        for (i = 0; i < An; ++i) {
            int row = An - 1 - i;
            if (c < row) { j = i + 1 + c; break; }
            c -= row;
        }
        float s = 0.0f;
        for (int k = 0; k < Dn; ++k) {
            float d = anchors[i * Dn + k] - anchors[j * Dn + k];
            s += d * d;
        }
        atomicAdd(&divacc, sqrtf(fmaxf(s, 0.0f)));
    }
    __syncthreads();
    if (t == 0) ws->div_sum = divacc;
}

// ---------------------------------------------------------------------------
// Kernel 2: precompute W_eff^T = (Wv @ proj_w)^T and w1^T, both bf16 N-major
// ---------------------------------------------------------------------------
__global__ void k_wprep(const float* __restrict__ qkv_w,
                        const float* __restrict__ proj_w,
                        const float* __restrict__ w1,
                        WS* __restrict__ ws) {
    const int tid = blockIdx.x * 256 + threadIdx.x;   // 65536 threads

    for (int o = tid; o < Dn * Dn; o += 65536) {
        const int n = o & (Dn - 1);
        const int k = o >> 9;
        float s = 0.0f;
        const float* __restrict__ wv_row = qkv_w + (size_t)k * (3 * Dn) + 2 * Dn;
        for (int m = 0; m < Dn; ++m)
            s += wv_row[m] * proj_w[(size_t)m * Dn + n];
        ws->Wt[(size_t)n * Dn + k] = (bf16)s;
    }
    for (int o = tid; o < HALF_D * Dn; o += 65536) {
        const int n = o >> 9;
        const int k = o & (Dn - 1);
        ws->w1T[o] = (bf16)w1[(size_t)k * HALF_D + n];
    }
}

// ---------------------------------------------------------------------------
// Kernel 3: main fused pass, 32 rows per workgroup, 8 waves (2 rb x 4 cg)
// ---------------------------------------------------------------------------
__global__ void __launch_bounds__(256)
k_main(const float* __restrict__ features,
       const float* __restrict__ w2,
       const float* __restrict__ b1,
       const float* __restrict__ b2,
       const WS* __restrict__ ws,
       float* __restrict__ refined_out,
       float* __restrict__ attn_out,
       float* __restrict__ center_sum_ptr) {
    // Fb doubles as the `balanced` bf16 tile for the final GEMM (features in
    // LDS are dead after the MLP phase; center loss re-reads global features).
    __shared__ __align__(16) bf16  Fb[ROWS][LDK];
    __shared__ __align__(16) float anch[An][Dn];
    __shared__ __align__(16) float H1[ROWS][LDH];
    __shared__ __align__(16) unsigned char Wpan[2][Dn * PANROW]; // TDM panels
    __shared__ float S[ROWS][12];
    __shared__ float attnw[ROWS][An];
    __shared__ float wgt[ROWS];
    __shared__ float h2[ROWS];
    __shared__ float cacc;

    const int t    = threadIdx.x;
    const int lane = t & 31;
    const int wid  = t >> 5;
    const int half = lane >> 4;
    const int l16  = lane & 15;
    const int rowbase = blockIdx.x * ROWS;

    if (t == 0) cacc = 0.0f;
    if (t < ROWS) h2[t] = 0.0f;
    for (int e = t; e < ROWS * 12; e += 256) S[e / 12][e % 12] = 0.0f;
    for (int e = t; e < An * Dn; e += 256) anch[e >> 9][e & 511] = ws->anch[e];

    // features tile -> LDS bf16
    for (int e = t * 4; e < ROWS * Dn; e += 256 * 4) {
        const float4 v = *(const float4*)&features[(size_t)rowbase * Dn + e];
        const int r = e >> 9, d = e & 511;
        Fb[r][d + 0] = (bf16)v.x; Fb[r][d + 1] = (bf16)v.y;
        Fb[r][d + 2] = (bf16)v.z; Fb[r][d + 3] = (bf16)v.w;
    }
    __syncthreads();

    // --- Phase 1: anchor dots + row norm -----------------------------------
    {
        const int r = t & 31, sl = t >> 5;
        float acc[An]; float nrm = 0.0f;
#pragma unroll
        for (int j = 0; j < An; ++j) acc[j] = 0.0f;
        for (int k = sl * 64; k < sl * 64 + 64; ++k) {
            const float f = (float)Fb[r][k];
            nrm += f * f;
#pragma unroll
            for (int j = 0; j < An; ++j) acc[j] += f * anch[j][k];
        }
#pragma unroll
        for (int j = 0; j < An; ++j) atomicAdd(&S[r][j], acc[j]);
        atomicAdd(&S[r][10], nrm);
    }
    __syncthreads();

    // softmax(-dist/||f||) per row
    if (t < ROWS) {
        const float nrm2 = S[t][10];
        const float temp = sqrtf(nrm2);
        float lg[An], mx = -1e30f;
#pragma unroll
        for (int j = 0; j < An; ++j) {
            const float d2 = nrm2 - 2.0f * S[t][j] + ws->anchnorm2[j];
            lg[j] = -sqrtf(fmaxf(d2, 0.0f)) / temp;
            mx = fmaxf(mx, lg[j]);
        }
        float sum = 0.0f;
#pragma unroll
        for (int j = 0; j < An; ++j) { lg[j] = __expf(lg[j] - mx); sum += lg[j]; }
        const float inv = 1.0f / sum;
#pragma unroll
        for (int j = 0; j < An; ++j) {
            const float a = lg[j] * inv;
            attnw[t][j] = a;
            attn_out[(size_t)(rowbase + t) * An + j] = a;
        }
    }

    // --- Phase 2: MLP GEMM  H1 = relu(F @ w1 + b1)  via WMMA ---------------
    // B-fragments preloaded as a batch (one clause + one wait per k-step).
    {
        const int rb = wid >> 2;
        const int cg = wid & 3;            // cols [cg*64, cg*64+64)
        v8f acc[4] = {};
        const bf16* __restrict__ w1T = ws->w1T;
        for (int kk = 0; kk < 16; ++kk) {
            const int k0 = kk * 32;
            BF16x16 a;
            a.h[0] = *(const v8bf*)&Fb[rb * 16 + l16][k0 + half * 8];
            a.h[1] = *(const v8bf*)&Fb[rb * 16 + l16][k0 + half * 8 + 16];
            BF16x16 b[4];
#pragma unroll
            for (int nb = 0; nb < 4; ++nb) {
                const int n = cg * 64 + nb * 16 + l16;
                b[nb].h[0] = *(const v8bf*)&w1T[(size_t)n * Dn + k0 + half * 8];
                b[nb].h[1] = *(const v8bf*)&w1T[(size_t)n * Dn + k0 + half * 8 + 16];
            }
#pragma unroll
            for (int nb = 0; nb < 4; ++nb)
                acc[nb] = __builtin_amdgcn_wmma_f32_16x16x32_bf16(
                    false, a.v, false, b[nb].v, (short)0, acc[nb], false, false);
        }
#pragma unroll
        for (int nb = 0; nb < 4; ++nb) {
            const int n = cg * 64 + nb * 16 + l16;
            const float bn = b1[n];
#pragma unroll
            for (int v = 0; v < 8; ++v)
                H1[rb * 16 + v + 8 * half][n] = fmaxf(acc[nb][v] + bn, 0.0f);
        }
    }
    __syncthreads();

    // h2 = H1 @ w2, gate = sigmoid(h2 + b2)
    {
        const int r = t & 31, sl = t >> 5;
        float p = 0.0f;
        for (int i = sl * 32; i < sl * 32 + 32; ++i) p += H1[r][i] * w2[i];
        atomicAdd(&h2[r], p);
    }
    __syncthreads();
    if (t < ROWS) wgt[t] = 1.0f / (1.0f + __expf(-(h2[t] + b2[0])));
    __syncthreads();

    // balanced = (attn @ anch) * gate  -> overwrite Fb (bf16)
    for (int e = t; e < ROWS * Dn; e += 256) {
        const int r = e >> 9, d = e & 511;
        float s = 0.0f;
#pragma unroll
        for (int j = 0; j < An; ++j) s += attnw[r][j] * anch[j][d];
        Fb[r][d] = (bf16)(s * wgt[r]);
    }
    __syncthreads();

    // --- Phase 3: refined = balanced @ W_eff + b_eff -----------------------
    // TDM streams Wt K-panels into double-buffered LDS; wave 0 drives the DMA
    // and TENSORcnt; barriers publish panels to the workgroup.
    {
        const int rb = wid >> 2;
        const int cg = wid & 3;            // cols [cg*128, cg*128+128)
        v8f acc[8] = {};
        const unsigned pan_off0 = (unsigned)(uintptr_t)&Wpan[0][0];
        const unsigned pan_off1 = (unsigned)(uintptr_t)&Wpan[1][0];

        if (wid == 0) tdm_load_panel(ws->Wt, pan_off0);     // prologue: panel 0

        for (int kk = 0; kk < 16; ++kk) {
            if (wid == 0) {
                if (kk < 15) {
                    tdm_load_panel(ws->Wt + (kk + 1) * 32,
                                   ((kk + 1) & 1) ? pan_off1 : pan_off0);
                    __builtin_amdgcn_s_wait_tensorcnt(1);   // panel kk done
                } else {
                    __builtin_amdgcn_s_wait_tensorcnt(0);
                }
            }
            __syncthreads();                                // publish panel kk

            const int k0 = kk * 32;
            const char* pan = (const char*)&Wpan[kk & 1][0];
            BF16x16 a;
            a.h[0] = *(const v8bf*)&Fb[rb * 16 + l16][k0 + half * 8];
            a.h[1] = *(const v8bf*)&Fb[rb * 16 + l16][k0 + half * 8 + 16];
            BF16x16 b[8];
#pragma unroll
            for (int nb = 0; nb < 8; ++nb) {
                const int n = cg * 128 + nb * 16 + l16;
                b[nb].h[0] = *(const v8bf*)(pan + n * PANROW + half * 16);
                b[nb].h[1] = *(const v8bf*)(pan + n * PANROW + half * 16 + 32);
            }
#pragma unroll
            for (int nb = 0; nb < 8; ++nb)
                acc[nb] = __builtin_amdgcn_wmma_f32_16x16x32_bf16(
                    false, a.v, false, b[nb].v, (short)0, acc[nb], false, false);

            __syncthreads();   // all waves done with buf[kk&1] before reuse
        }

        // epilogue: + b_eff, store refined, accumulate center loss
        float cpart = 0.0f;
#pragma unroll
        for (int nb = 0; nb < 8; ++nb) {
            const int n = cg * 128 + nb * 16 + l16;
            const float be = ws->b_eff[n];
#pragma unroll
            for (int v = 0; v < 8; ++v) {
                const int r = rowbase + rb * 16 + v + 8 * half;
                const float val = acc[nb][v] + be;
                refined_out[(size_t)r * Dn + n] = val;
                const float d = features[(size_t)r * Dn + n] - val;
                cpart += d * d;
            }
        }
        atomicAdd(&cacc, cpart);
    }
    __syncthreads();
    if (t == 0) atomicAdd(center_sum_ptr, cacc);
}

// ---------------------------------------------------------------------------
// Kernel 4: scalar loss
// ---------------------------------------------------------------------------
__global__ void k_final(const WS* __restrict__ ws,
                        const float* __restrict__ clw,
                        float* __restrict__ out_loss) {
    const float center = ws->center_sum / (float)Bn;
    const float div = -(ws->div_sum / (float)((An * (An - 1)) / 2));
    *out_loss = clw[0] * center + 0.1f * div;
}

// ---------------------------------------------------------------------------
extern "C" void kernel_launch(void* const* d_in, const int* in_sizes, int n_in,
                              void* d_out, int out_size, void* d_ws, size_t ws_size,
                              hipStream_t stream) {
    (void)in_sizes; (void)n_in; (void)out_size; (void)ws_size;
    const float* features = (const float*)d_in[0];
    const float* anchors  = (const float*)d_in[1];
    const float* pos      = (const float*)d_in[2];
    const float* qkv_w    = (const float*)d_in[3];
    const float* qkv_b    = (const float*)d_in[4];
    const float* proj_w   = (const float*)d_in[5];
    const float* proj_b   = (const float*)d_in[6];
    const float* w1       = (const float*)d_in[7];
    const float* b1       = (const float*)d_in[8];
    const float* w2       = (const float*)d_in[9];
    const float* b2       = (const float*)d_in[10];
    const float* clw      = (const float*)d_in[11];

    float* out     = (float*)d_out;
    float* refined = out;
    float* loss    = out + (size_t)Bn * Dn;
    float* attn    = loss + 1;
    WS* ws = (WS*)d_ws;

    k_prep <<<1,   256, 0, stream>>>(anchors, pos, qkv_b, proj_w, proj_b, ws);
    k_wprep<<<256, 256, 0, stream>>>(qkv_w, proj_w, w1, ws);
    k_main <<<Bn / ROWS, 256, 0, stream>>>(features, w2, b1, b2, ws,
                                           refined, attn, &ws->center_sum);
    k_final<<<1, 1, 0, stream>>>(ws, clw, loss);
}
